// Half_Graph_77086073029171
// MI455X (gfx1250) — matrix operand order repaired and checked by reading, not compiled, not run on hardware
//
#include <hip/hip_runtime.h>

typedef __attribute__((ext_vector_type(16))) _Float16 v16h;
typedef __attribute__((ext_vector_type(8)))  _Float16 v8h;
typedef __attribute__((ext_vector_type(8)))  float    v8f;

static constexpr int BN  = 8;          // batch
static constexpr int HH  = 96;
static constexpr int WW  = 96;
static constexpr int SP  = HH * WW;    // 9216 pixels per plane
static constexpr int HP  = 98;         // haloed spatial dim
static constexpr int CP  = 288;        // 257 input channels padded to 9 chunks of 32
static constexpr int NKC = 9;          // K-chunks per tap
static constexpr int NOUTC = 256;

// ---------------- workspace layout (bytes) ----------------
static constexpr size_t SZ_X    = (size_t)BN * HP * HP * CP * 2;      // 44,255,232
static constexpr size_t SZ_W    = (size_t)9 * NKC * NOUTC * 32 * 2;   //  1,327,104
static constexpr size_t SZ_DATT = (size_t)BN * 2 * SP * 4;            //    589,824
static constexpr size_t SZ_BRIN = (size_t)4 * BN * 20 * SP * 4;       // 23,592,960
static constexpr size_t SZ_BRMD = (size_t)4 * BN * 10 * SP * 4;       // 11,796,480

static constexpr size_t OFF_X    = 0;
static constexpr size_t OFF_W    = OFF_X + SZ_X;
static constexpr size_t OFF_DATT = OFF_W + SZ_W;
static constexpr size_t OFF_BRIN = OFF_DATT + SZ_DATT;
static constexpr size_t OFF_BRMD = OFF_BRIN + SZ_BRIN;
static constexpr size_t OFF_BROUT= OFF_BRMD + SZ_BRMD;
// total required: OFF_BROUT + SZ_BRMD = 93,358,080 bytes (~89 MB)

// ---------------- kernel 1: pack [parent_att, xh] -> f16 haloed, channel-padded ----------------
__global__ void pack_x16(const float* __restrict__ xh, const float* __restrict__ f_atts,
                         _Float16* __restrict__ X16, int total)
{
    int idx = blockIdx.x * blockDim.x + threadIdx.x;
    if (idx >= total) return;
    int c  = idx % CP;
    int p  = idx / CP;
    int xw = p % HP;
    int yw = (p / HP) % HP;
    int b  = p / (HP * HP);
    float v = 0.f;
    if (xw >= 1 && xw <= WW && yw >= 1 && yw <= HH) {
        int sp = (yw - 1) * WW + (xw - 1);
        if (c == 0)        v = f_atts[(size_t)(1 * BN + b) * SP + sp];       // f_atts[1][b][0]
        else if (c <= 256) v = xh[((size_t)b * 256 + (c - 1)) * SP + sp];
    }
    X16[idx] = (_Float16)v;
}

// ---------------- kernel 2: repack da_w1 [256,257,3,3] -> [tap][kc][n][k32] f16 ----------------
__global__ void pack_wda(const float* __restrict__ w1, _Float16* __restrict__ WB, int total)
{
    int idx = blockIdx.x * blockDim.x + threadIdx.x;
    if (idx >= total) return;
    int k   = idx % 32;
    int n   = (idx / 32) % NOUTC;
    int kc  = (idx / (32 * NOUTC)) % NKC;
    int tap = idx / (32 * NOUTC * NKC);
    int c   = kc * 32 + k;
    float v = 0.f;
    if (c < 257) {
        int ty = tap / 3, tx = tap % 3;
        v = w1[(((size_t)n * 257 + c) * 3 + ty) * 3 + tx];
    }
    WB[idx] = (_Float16)v;
}

// ---------------- kernel 3: WMMA implicit-GEMM 3x3 conv (257->256) -------------------
// Block = 64 pixels x 256 output channels; 8 waves as 2 m-groups x 4 n-groups;
// each wave holds 2 m-tiles x 4 n-tiles (8 accumulators) for register-level B reuse:
// 12 B128 loads feed 8 WMMAs per K-step. Epilogue fuses BN+ReLU+1x1(256->2)+softmax.
__global__ void __launch_bounds__(256)
conv_da_wmma(const _Float16* __restrict__ X16, const _Float16* __restrict__ WB,
             const float* __restrict__ g1, const float* __restrict__ b1,
             const float* __restrict__ w2, const float* __restrict__ b2,
             float* __restrict__ out_dm, float* __restrict__ datt)
{
    __shared__ _Float16 lds_y1[64 * 256];   // 64 pixels x 256 channels, f16 = 32 KB
    const int tid  = threadIdx.x;
    const int wv   = tid >> 5;           // wave id, 0..7 (wave32)
    const int lane = tid & 31;
    const int nl   = lane & 15;
    const int hi   = lane >> 4;
    const int mg   = wv >> 2;            // m-group 0..1 (each owns 2 m-tiles)
    const int ng   = wv & 3;             // n-group 0..3 (each owns 64 channels)

    const int p0  = blockIdx.x * 64;     // 64 pixels; each 16-aligned sub-tile stays in one row
    const int nb  = ng * 64;             // this wave's channel base

    // per-wave m-tiles: (b,y,x) of this lane's pixel in each tile
    int pb[2], py[2], pxm[2];
    #pragma unroll
    for (int mi = 0; mi < 2; ++mi) {
        int pt = p0 + (mg * 2 + mi) * 16;
        pb[mi]  = pt / SP;
        py[mi]  = (pt / WW) % HH;
        pxm[mi] = pt % WW + nl;          // x0 + m
    }

    v8f acc[2][4];
    #pragma unroll
    for (int mi = 0; mi < 2; ++mi)
        #pragma unroll
        for (int ni = 0; ni < 4; ++ni)
            acc[mi][ni] = {};

    for (int tap = 0; tap < 9; ++tap) {
        const int dy = tap / 3, dx = tap % 3;
        const _Float16* arow0 = X16 + (size_t)(((pb[0] * HP) + (py[0] + dy)) * HP + (pxm[0] + dx)) * CP;
        const _Float16* arow1 = X16 + (size_t)(((pb[1] * HP) + (py[1] + dy)) * HP + (pxm[1] + dx)) * CP;
        const _Float16* wtap  = WB + (size_t)(tap * NKC) * NOUTC * 32;
        #pragma unroll
        for (int kc = 0; kc < NKC; ++kc) {
            // A fragments: lanes 0-15 hold K=0..15; lanes 16-31 hold K=16..31
            v16h a0, a1;
            ((v8h*)&a0)[0] = *(const v8h*)(arow0 + kc * 32 + hi * 16);
            ((v8h*)&a0)[1] = *(const v8h*)(arow0 + kc * 32 + hi * 16 + 8);
            ((v8h*)&a1)[0] = *(const v8h*)(arow1 + kc * 32 + hi * 16);
            ((v8h*)&a1)[1] = *(const v8h*)(arow1 + kc * 32 + hi * 16 + 8);
            // B fragments: lane = (n = l%16, K-run hi*16..+15), contiguous in packed weights
            const _Float16* wk = wtap + (size_t)kc * NOUTC * 32;
            v16h bf[4];
            #pragma unroll
            for (int ni = 0; ni < 4; ++ni) {
                const _Float16* wp = wk + (nb + ni * 16 + nl) * 32 + hi * 16;
                ((v8h*)&bf[ni])[0] = *(const v8h*)(wp);
                ((v8h*)&bf[ni])[1] = *(const v8h*)(wp + 8);
            }
            #pragma unroll
            for (int ni = 0; ni < 4; ++ni) {
                acc[0][ni] = __builtin_amdgcn_wmma_f32_16x16x32_f16(false, a0, false, bf[ni],
                                                                    (short)0, acc[0][ni], false, false);
                acc[1][ni] = __builtin_amdgcn_wmma_f32_16x16x32_f16(false, a1, false, bf[ni],
                                                                    (short)0, acc[1][ni], false, false);
            }
        }
    }

    // BN + ReLU, stage y1 (f16) into LDS (C/D layout: VGPR v -> row v + 8*hi, col = nl)
    #pragma unroll
    for (int mi = 0; mi < 2; ++mi) {
        int mrow = (mg * 2 + mi) * 16;
        #pragma unroll
        for (int ni = 0; ni < 4; ++ni) {
            int n = nb + ni * 16 + nl;
            float g = g1[n], bb = b1[n];
            #pragma unroll
            for (int v = 0; v < 8; ++v) {
                int mm = mrow + v + 8 * hi;
                lds_y1[mm * 256 + n] = (_Float16)fmaxf(acc[mi][ni][v] * g + bb, 0.f);
            }
        }
    }
    __syncthreads();

    // fused 1x1 conv (256->2) + bias + softmax over the 2 channels
    if (tid < 64) {
        int pm = p0 + tid;
        int b  = pm / SP;
        int y  = (pm / WW) % HH;
        int x  = pm % WW;
        float s0 = 0.f, s1 = 0.f;
        for (int n = 0; n < 256; ++n) {
            float v = (float)lds_y1[tid * 256 + n];
            s0 += v * w2[n];
            s1 += v * w2[256 + n];
        }
        float d0 = s0 + b2[0], d1 = s1 + b2[1];
        float md = fmaxf(d0, d1);
        float e0 = __expf(d0 - md), e1 = __expf(d1 - md);
        float inv = 1.f / (e0 + e1);
        size_t o0 = (size_t)((b * 2 + 0) * HH + y) * WW + x;
        size_t o1 = (size_t)((b * 2 + 1) * HH + y) * WW + x;
        out_dm[o0] = d0;       out_dm[o1] = d1;
        datt[o0]   = e0 * inv; datt[o1]   = e1 * inv;
    }
}

// ---------------- kernel 4: build the four branch inputs (20 channels each) ----------------
__global__ void mk_branch_in(const float* __restrict__ f_nodes, const float* __restrict__ h_nodes,
                             const float* __restrict__ p_nodes, const float* __restrict__ f_atts,
                             const float* __restrict__ p_atts, const float* __restrict__ datt,
                             float* __restrict__ br_in, int total)
{
    int idx = blockIdx.x * blockDim.x + threadIdx.x;
    if (idx >= total) return;
    int sp = idx % SP;
    int c  = (idx / SP) % 20;
    int b  = (idx / (SP * 20)) % BN;
    int j  = idx / (SP * 20 * BN);     // 0:rel0 1:rel1 2:comp_u 3:comp_l
    float v;
    if (j < 2) {
        if (c < 10) {
            float par = f_nodes[((size_t)(1 * BN + b) * 10 + c) * SP + sp];
            float da  = datt[(size_t)(b * 2 + j) * SP + sp];
            float pa  = f_atts[(size_t)(1 * BN + b) * SP + sp];
            v = par * da * pa;
        } else {
            v = h_nodes[((size_t)((1 + j) * BN + b) * 10 + (c - 10)) * SP + sp];
        }
    } else if (j == 2) {
        if (c < 10) v = h_nodes[((size_t)(1 * BN + b) * 10 + c) * SP + sp];
        else {
            float sn = 0.f, sa = 0.f;
            for (int k = 1; k <= 4; ++k) {
                sn += p_nodes[((size_t)(k * BN + b) * 10 + (c - 10)) * SP + sp];
                sa += p_atts[(size_t)(k * BN + b) * SP + sp];
            }
            v = sn * sa;
        }
    } else {
        if (c < 10) v = h_nodes[((size_t)(2 * BN + b) * 10 + c) * SP + sp];
        else {
            float sn = 0.f, sa = 0.f;
            for (int k = 5; k <= 6; ++k) {
                sn += p_nodes[((size_t)(k * BN + b) * 10 + (c - 10)) * SP + sp];
                sa += p_atts[(size_t)(k * BN + b) * SP + sp];
            }
            v = sn * sa;
        }
    }
    br_in[idx] = v;
}

// ---------------- kernel 5: branch 3x3 conv (20->10) + BN + ReLU, LDS-tiled ----------------
__global__ void __launch_bounds__(256)
branch_conv3(const float* __restrict__ br_in,
             const float* __restrict__ rw1, const float* __restrict__ rg1, const float* __restrict__ rb1,
             const float* __restrict__ uw1, const float* __restrict__ ug1, const float* __restrict__ ub1,
             const float* __restrict__ lw1, const float* __restrict__ lg1, const float* __restrict__ lb1,
             float* __restrict__ br_mid)
{
    __shared__ float s_in[20][18][18];
    __shared__ float s_w[1800];
    int z = blockIdx.z;               // j*BN + b
    int j = z / BN, b = z % BN;
    const float* w1 = (j < 2) ? rw1 : (j == 2 ? uw1 : lw1);
    const float* g1 = (j < 2) ? rg1 : (j == 2 ? ug1 : lg1);
    const float* b1 = (j < 2) ? rb1 : (j == 2 ? ub1 : lb1);
    int tx0 = blockIdx.x * 16, ty0 = blockIdx.y * 16;
    int tid = threadIdx.y * 16 + threadIdx.x;
    const float* inb = br_in + (size_t)(j * BN + b) * 20 * SP;

    for (int i = tid; i < 20 * 18 * 18; i += 256) {
        int lx = i % 18, ly = (i / 18) % 18, c = i / (18 * 18);
        int gx = tx0 + lx - 1, gy = ty0 + ly - 1;
        float v = 0.f;
        if (gx >= 0 && gx < WW && gy >= 0 && gy < HH)
            v = inb[(size_t)c * SP + gy * WW + gx];
        s_in[c][ly][lx] = v;
    }
    for (int i = tid; i < 1800; i += 256) s_w[i] = w1[i];
    __syncthreads();

    int lx = threadIdx.x, ly = threadIdx.y;
    float acc[10];
    #pragma unroll
    for (int o = 0; o < 10; ++o) acc[o] = 0.f;
    for (int c = 0; c < 20; ++c) {
        float vv[9];
        #pragma unroll
        for (int ky = 0; ky < 3; ++ky)
            #pragma unroll
            for (int kx = 0; kx < 3; ++kx)
                vv[ky * 3 + kx] = s_in[c][ly + ky][lx + kx];
        #pragma unroll
        for (int o = 0; o < 10; ++o) {
            const float* wo = &s_w[(o * 20 + c) * 9];
            #pragma unroll
            for (int k = 0; k < 9; ++k) acc[o] += vv[k] * wo[k];
        }
    }
    float* outb = br_mid + (size_t)(j * BN + b) * 10 * SP;
    int x = tx0 + lx, y = ty0 + ly;
    #pragma unroll
    for (int o = 0; o < 10; ++o)
        outb[(size_t)o * SP + y * WW + x] = fmaxf(acc[o] * g1[o] + b1[o], 0.f);
}

// ---------------- kernel 6: branch 1x1 conv (10->10) + BN + ReLU ----------------
__global__ void branch_1x1(const float* __restrict__ br_mid,
                           const float* __restrict__ rw2, const float* __restrict__ rg2, const float* __restrict__ rb2,
                           const float* __restrict__ uw2, const float* __restrict__ ug2, const float* __restrict__ ub2,
                           const float* __restrict__ lw2, const float* __restrict__ lg2, const float* __restrict__ lb2,
                           float* __restrict__ br_out, int total)
{
    int idx = blockIdx.x * blockDim.x + threadIdx.x;
    if (idx >= total) return;
    int sp = idx % SP;
    int b  = (idx / SP) % BN;
    int j  = idx / (SP * BN);
    const float* w2 = (j < 2) ? rw2 : (j == 2 ? uw2 : lw2);
    const float* g2 = (j < 2) ? rg2 : (j == 2 ? ug2 : lg2);
    const float* b2 = (j < 2) ? rb2 : (j == 2 ? ub2 : lb2);
    const float* inb = br_mid + (size_t)(j * BN + b) * 10 * SP + sp;
    float xin[10];
    #pragma unroll
    for (int c = 0; c < 10; ++c) xin[c] = inb[(size_t)c * SP];
    float* outb = br_out + (size_t)(j * BN + b) * 10 * SP + sp;
    #pragma unroll
    for (int o = 0; o < 10; ++o) {
        float a = 0.f;
        #pragma unroll
        for (int c = 0; c < 10; ++c) a += xin[c] * w2[o * 10 + c];
        outb[(size_t)o * SP] = fmaxf(a * g2[o] + b2[o], 0.f);
    }
}

// ---------------- kernel 7: fused ConvGRU (1x1) for the 3 nodes ----------------
__global__ void gru_kernel(const float* __restrict__ f_nodes, const float* __restrict__ h_nodes,
                           const float* __restrict__ p_nodes, const float* __restrict__ br_out,
                           const float* __restrict__ gw, const float* __restrict__ gb,
                           const float* __restrict__ cw, const float* __restrict__ cg,
                           const float* __restrict__ cb, float* __restrict__ out, int total)
{
    int idx = blockIdx.x * blockDim.x + threadIdx.x;
    if (idx >= total) return;
    int sp = idx % SP;
    int b  = (idx / SP) % BN;
    int nd = idx / (SP * BN);
    float xin[10], h[10];
    if (nd == 0) {
        #pragma unroll
        for (int c = 0; c < 10; ++c)
            xin[c] = f_nodes[((size_t)b * 10 + c) * SP + sp] + p_nodes[((size_t)b * 10 + c) * SP + sp];
    } else {
        int ja = (nd == 1) ? 2 : 3;   // comp_u / comp_l
        int jb = (nd == 1) ? 0 : 1;   // decomp[0] / decomp[1]
        #pragma unroll
        for (int c = 0; c < 10; ++c)
            xin[c] = br_out[((size_t)(ja * BN + b) * 10 + c) * SP + sp]
                   + br_out[((size_t)(jb * BN + b) * 10 + c) * SP + sp];
    }
    #pragma unroll
    for (int c = 0; c < 10; ++c)
        h[c] = h_nodes[((size_t)(nd * BN + b) * 10 + c) * SP + sp];

    const float* gwn = gw + nd * 40;   // [2][20]
    const float* cwn = cw + nd * 200;  // [10][20]
    float z0 = gb[nd * 2 + 0], z1 = gb[nd * 2 + 1];
    #pragma unroll
    for (int c = 0; c < 10; ++c) {
        z0 += xin[c] * gwn[c] + h[c] * gwn[10 + c];
        z1 += xin[c] * gwn[20 + c] + h[c] * gwn[30 + c];
    }
    float r = 1.f / (1.f + __expf(-z0));
    float u = 1.f / (1.f + __expf(-z1));
    #pragma unroll
    for (int o = 0; o < 10; ++o) {
        float a = 0.f;
        #pragma unroll
        for (int c = 0; c < 10; ++c)
            a += xin[c] * cwn[o * 20 + c] + (r * h[c]) * cwn[o * 20 + 10 + c];
        float cnm = fmaxf(a * cg[nd * 10 + o] + cb[nd * 10 + o], 0.f);
        out[((size_t)(nd * BN + b) * 10 + o) * SP + sp] = (1.f - u) * h[o] + u * cnm;
    }
}

extern "C" void kernel_launch(void* const* d_in, const int* in_sizes, int n_in,
                              void* d_out, int out_size, void* d_ws, size_t ws_size,
                              hipStream_t stream)
{
    (void)in_sizes; (void)n_in; (void)out_size; (void)ws_size;
    const float* f_nodes = (const float*)d_in[0];
    const float* h_nodes = (const float*)d_in[1];
    const float* p_nodes = (const float*)d_in[2];
    const float* xh      = (const float*)d_in[3];
    const float* f_atts  = (const float*)d_in[4];
    const float* p_atts  = (const float*)d_in[5];
    const float* da_w1 = (const float*)d_in[6];
    const float* da_g1 = (const float*)d_in[7];
    const float* da_b1 = (const float*)d_in[8];
    const float* da_w2 = (const float*)d_in[9];
    const float* da_b2 = (const float*)d_in[10];
    const float* rel_w1 = (const float*)d_in[11];
    const float* rel_g1 = (const float*)d_in[12];
    const float* rel_b1 = (const float*)d_in[13];
    const float* rel_w2 = (const float*)d_in[14];
    const float* rel_g2 = (const float*)d_in[15];
    const float* rel_b2 = (const float*)d_in[16];
    const float* cu_w1 = (const float*)d_in[17];
    const float* cu_g1 = (const float*)d_in[18];
    const float* cu_b1 = (const float*)d_in[19];
    const float* cu_w2 = (const float*)d_in[20];
    const float* cu_g2 = (const float*)d_in[21];
    const float* cu_b2 = (const float*)d_in[22];
    const float* cl_w1 = (const float*)d_in[23];
    const float* cl_g1 = (const float*)d_in[24];
    const float* cl_b1 = (const float*)d_in[25];
    const float* cl_w2 = (const float*)d_in[26];
    const float* cl_g2 = (const float*)d_in[27];
    const float* cl_b2 = (const float*)d_in[28];
    const float* gru_gw = (const float*)d_in[29];
    const float* gru_gb = (const float*)d_in[30];
    const float* gru_cw = (const float*)d_in[31];
    const float* gru_cg = (const float*)d_in[32];
    const float* gru_cb = (const float*)d_in[33];

    char* ws = (char*)d_ws;
    _Float16* X16  = (_Float16*)(ws + OFF_X);
    _Float16* WB   = (_Float16*)(ws + OFF_W);
    float* datt    = (float*)(ws + OFF_DATT);
    float* br_in   = (float*)(ws + OFF_BRIN);
    float* br_mid  = (float*)(ws + OFF_BRMD);
    float* br_out  = (float*)(ws + OFF_BROUT);
    float* out_nodes = (float*)d_out;
    float* out_dm    = out_nodes + (size_t)3 * BN * 10 * SP;

    { int n = BN * HP * HP * CP;
      pack_x16<<<(n + 255) / 256, 256, 0, stream>>>(xh, f_atts, X16, n); }
    { int n = 9 * NKC * NOUTC * 32;
      pack_wda<<<(n + 255) / 256, 256, 0, stream>>>(da_w1, WB, n); }

    conv_da_wmma<<<(BN * SP) / 64, 256, 0, stream>>>(X16, WB, da_g1, da_b1, da_w2, da_b2,
                                                     out_dm, datt);

    { int n = 4 * BN * 20 * SP;
      mk_branch_in<<<(n + 255) / 256, 256, 0, stream>>>(f_nodes, h_nodes, p_nodes,
                                                        f_atts, p_atts, datt, br_in, n); }

    branch_conv3<<<dim3(WW / 16, HH / 16, 4 * BN), dim3(16, 16), 0, stream>>>(
        br_in, rel_w1, rel_g1, rel_b1, cu_w1, cu_g1, cu_b1, cl_w1, cl_g1, cl_b1, br_mid);

    { int n = 4 * BN * SP;
      branch_1x1<<<(n + 255) / 256, 256, 0, stream>>>(br_mid, rel_w2, rel_g2, rel_b2,
                                                      cu_w2, cu_g2, cu_b2, cl_w2, cl_g2, cl_b2,
                                                      br_out, n); }

    { int n = 3 * BN * SP;
      gru_kernel<<<(n + 255) / 256, 256, 0, stream>>>(f_nodes, h_nodes, p_nodes, br_out,
                                                      gru_gw, gru_gb, gru_cw, gru_cg, gru_cb,
                                                      out_nodes, n); }
}
